// WhisperToxicSpansKDModel_79783312490692
// MI455X (gfx1250) — compile-verified, matrix operand order checked by reading, not run on hardware
//
#include <hip/hip_runtime.h>
#include <hip/hip_bf16.h>

typedef unsigned short u16;
typedef unsigned int u32;
typedef __attribute__((ext_vector_type(16))) __bf16 v16bf;
typedef __attribute__((ext_vector_type(8)))  float  v8f;

#define BB 32
#define TT 448
#define DD 768
#define HH 256
#define G4 1024            // 4*H
#define MM (BB*TT)         // 14336 rows of the input GEMM
#define KD_BLOCKS 1024

// ---------------- helpers ----------------

__device__ __forceinline__ u16 f2bf(float f) {            // RNE f32 -> bf16
  u32 x = __float_as_uint(f);
  u32 r = x + 0x7FFFu + ((x >> 16) & 1u);
  return (u16)(r >> 16);
}
__device__ __forceinline__ float sigf(float x) { return 1.0f / (1.0f + __expf(-x)); }
__device__ __forceinline__ float lse2(float a, float b) {
  float m = fmaxf(a, b);
  return m + __logf(__expf(a - m) + __expf(b - m));
}

union FragAB { v16bf v; uint4 q[2]; };

// Row-major bf16 matrix (rows x ldk). Loads the 16x32 A-style fragment
// (also valid as B-fragment of W^T when row = output column n).
__device__ __forceinline__ FragAB load_frag(const u16* __restrict__ base,
                                            int row, int ldk, int k0, int lane) {
  FragAB f;
  const int kofs = (lane >> 4) * 8;
  const u16* p = base + (size_t)row * ldk + k0 + kofs;
  f.q[0] = *(const uint4*)(p);        // K = k0+kofs   .. +7
  f.q[1] = *(const uint4*)(p + 16);   // K = k0+16+kofs.. +7
  return f;
}

// ---------------- f32 -> bf16 convert ----------------

__global__ void cvt_bf16_kernel(const float* __restrict__ src, u16* __restrict__ dst, size_t n) {
  size_t i = (size_t)blockIdx.x * blockDim.x + threadIdx.x;
  size_t stride = (size_t)gridDim.x * blockDim.x;
  for (; i < n; i += stride) dst[i] = f2bf(src[i]);
}

// ---------------- KD loss partial reduction ----------------

__global__ void kd_kernel(const float* __restrict__ s, const float* __restrict__ t,
                          float* __restrict__ part, size_t n) {
  __shared__ float wsum[8];
  size_t i = (size_t)blockIdx.x * blockDim.x + threadIdx.x;
  size_t stride = (size_t)gridDim.x * blockDim.x;
  float acc = 0.f;
  for (; i < n; i += stride) { float d = s[i] - t[i]; acc += d * d; }
  for (int o = 16; o >= 1; o >>= 1) acc += __shfl_xor(acc, o, 32);
  const int lane = threadIdx.x & 31, wid = threadIdx.x >> 5;
  if (lane == 0) wsum[wid] = acc;
  __syncthreads();
  if (wid == 0) {
    float v = (lane < 8) ? wsum[lane] : 0.f;
    for (int o = 4; o >= 1; o >>= 1) v += __shfl_xor(v, o, 32);
    if (lane == 0) part[blockIdx.x] = v;
  }
}

// ---------------- gate pre-activation GEMM: gx = x @ W_ih^T + b ----------------
// A: x_bf16 (14336 x 768) row-major.  B: W_ih (1024 x 768) row-major == (K x N)^T.
// Each wave computes a 16x64 strip (4 N-tiles) so one A fragment feeds 4 WMMAs:
// steady state is 10 x b128 loads -> 4 x v_wmma per K-step.

__global__ __launch_bounds__(256) void gx_gemm_kernel(
    const u16* __restrict__ xb,
    const u16* __restrict__ wf, const u16* __restrict__ wb,
    const float* __restrict__ bihf, const float* __restrict__ bhhf,
    const float* __restrict__ bihb, const float* __restrict__ bhhb,
    float* __restrict__ gxf, float* __restrict__ gxb)
{
  const int dir = blockIdx.z;
  const u16*  w  = dir ? wb   : wf;
  const float* bi = dir ? bihb : bihf;
  const float* bh = dir ? bhhb : bhhf;
  float* gx = dir ? gxb : gxf;

  const int lane = threadIdx.x & 31;
  const int wid  = threadIdx.x >> 5;
  const int task = blockIdx.x * 8 + wid;   // 14336 wave tasks / dir
  const int mt   = task >> 4;              // 896 M-tiles
  const int ngrp = task & 15;              // 16 groups x 4 N-tiles
  const int m0   = mt * 16;

  const int arow = m0 + (lane & 15);
  const int kofs = (lane >> 4) * 8;

  v8f acc[4] = {};
  for (int k0 = 0; k0 < DD; k0 += 32) {
    FragAB a = load_frag(xb, arow, DD, k0, lane);
    if (k0 + 32 < DD)   // early-translation hint for next A block (global_prefetch_b8)
      __builtin_prefetch(xb + (size_t)arow * DD + k0 + 32 + kofs, 0, 1);
#pragma unroll
    for (int i = 0; i < 4; ++i) {
      const int brow = (ngrp * 4 + i) * 16 + (lane & 15);
      FragAB b = load_frag(w, brow, DD, k0, lane);
      acc[i] = __builtin_amdgcn_wmma_f32_16x16x32_bf16(false, a.v, false, b.v,
                                                       (short)0, acc[i], false, false);
    }
  }

  const int rbase = m0 + ((lane >> 4) * 8);
#pragma unroll
  for (int i = 0; i < 4; ++i) {
    const int col = (ngrp * 4 + i) * 16 + (lane & 15);
    const float bias = bi[col] + bh[col];
#pragma unroll
    for (int r = 0; r < 8; ++r)
      gx[(size_t)(rbase + r) * G4 + col] = acc[i][r] + bias;
  }
}

// ---------------- persistent LSTM scan (one workgroup per direction) ----------------
// LDS: g (32x1024 f32, 128KB) + c (32x256 f32, 32KB) + h_bf16 (32x256, 16KB) = 176KB
// of gfx1250's 320KB/WGP. Per step: g = h @ Whh^T (WMMA, A from LDS via ds_load_b128)
// + gx[:,t,:]; then fused sigmoid/tanh gate update in LDS.

__global__ __launch_bounds__(1024) void lstm_scan_kernel(
    const float* __restrict__ gx_f, const float* __restrict__ gx_b,
    const u16* __restrict__ whh_f, const u16* __restrict__ whh_b,
    float* __restrict__ hs_f, float* __restrict__ hs_b)
{
  extern __shared__ char smem[];
  float* g_s  = (float*)smem;                                   // 32 x 1024
  float* c_s  = (float*)(smem + (size_t)BB * G4 * 4);           // 32 x 256
  u16*   hb_s = (u16*)  (smem + (size_t)BB * G4 * 4 + (size_t)BB * HH * 4); // 32 x 256

  const int dir = blockIdx.x;
  const float* gx = dir ? gx_b : gx_f;
  const u16* whh  = dir ? whh_b : whh_f;
  float* hs       = dir ? hs_b : hs_f;

  const int tid  = threadIdx.x;
  const int lane = tid & 31;
  const int wid  = tid >> 5;

  for (int i = tid; i < BB * HH; i += 1024) { c_s[i] = 0.f; hb_s[i] = 0; }
  __syncthreads();

  const int mt   = wid >> 4;           // 2 M-tiles (batch 0..15 / 16..31)
  const int ngrp = wid & 15;           // 16 groups x 4 N-tiles = 64 N-tiles
  const int arow = mt * 16 + (lane & 15);
  const int kofs = (lane >> 4) * 8;

  for (int step = 0; step < TT; ++step) {
    const int t = dir ? (TT - 1 - step) : step;

    v8f acc[4] = {};
    for (int k0 = 0; k0 < HH; k0 += 32) {
      FragAB a;
      const u16* pa = hb_s + arow * HH + k0 + kofs;
      a.q[0] = *(const uint4*)(pa);
      a.q[1] = *(const uint4*)(pa + 16);
#pragma unroll
      for (int i = 0; i < 4; ++i) {
        const int nt = ngrp * 4 + i;
        FragAB b = load_frag(whh, nt * 16 + (lane & 15), HH, k0, lane);
        acc[i] = __builtin_amdgcn_wmma_f32_16x16x32_bf16(false, a.v, false, b.v,
                                                         (short)0, acc[i], false, false);
      }
    }

    const int rbase = mt * 16 + ((lane >> 4) * 8);
#pragma unroll
    for (int i = 0; i < 4; ++i) {
      const int col = (ngrp * 4 + i) * 16 + (lane & 15);
#pragma unroll
      for (int r = 0; r < 8; ++r) {
        const int m = rbase + r;
        g_s[m * G4 + col] = acc[i][r] + gx[((size_t)m * TT + t) * G4 + col];
      }
    }
    __syncthreads();

    // fused gate update: i,f,g,o at cols j, 256+j, 512+j, 768+j
    for (int u = tid; u < BB * HH; u += 1024) {
      const int m = u >> 8, j = u & 255;
      const float gi = g_s[m * G4 + j];
      const float gf = g_s[m * G4 + HH + j];
      const float gg = g_s[m * G4 + 2 * HH + j];
      const float go = g_s[m * G4 + 3 * HH + j];
      const float cn = sigf(gf) * c_s[u] + sigf(gi) * tanhf(gg);
      const float h  = sigf(go) * tanhf(cn);
      c_s[u]  = cn;
      hb_s[u] = f2bf(h);
      hs[((size_t)m * TT + t) * HH + j] = h;
    }
    __syncthreads();
  }
}

// ---------------- classifier + CRF (one wave per batch row) ----------------

__global__ __launch_bounds__(32) void crf_kernel(
    const float* __restrict__ hs_f, const float* __restrict__ hs_b,
    const float* __restrict__ W_cls, const float* __restrict__ b_cls,
    const float* __restrict__ crf_start, const float* __restrict__ crf_end,
    const float* __restrict__ crf_trans,
    const int* __restrict__ labels, const int* __restrict__ mask,
    float* __restrict__ res)
{
  const int b = blockIdx.x;
  const int lane = threadIdx.x;

  float w0[16], w1[16];
#pragma unroll
  for (int i = 0; i < 16; ++i) {
    w0[i] = W_cls[lane * 16 + i];
    w1[i] = W_cls[512 + lane * 16 + i];
  }
  const bool useF = lane < 16;
  const int jbase = useF ? lane * 16 : lane * 16 - HH;
  const float* hsrc = useF ? hs_f : hs_b;

  const float t00 = crf_trans[0], t01 = crf_trans[1], t10 = crf_trans[2], t11 = crf_trans[3];
  float alpha0 = 0.f, alpha1 = 0.f, num = 0.f;
  int prev_tag = 0;

  for (int t = 0; t < TT; ++t) {
    const float* hp = hsrc + ((size_t)b * TT + t) * HH + jbase;
    float p0 = 0.f, p1 = 0.f;
#pragma unroll
    for (int i = 0; i < 16; ++i) { const float h = hp[i]; p0 += w0[i] * h; p1 += w1[i] * h; }
    for (int o = 16; o >= 1; o >>= 1) { p0 += __shfl_xor(p0, o, 32); p1 += __shfl_xor(p1, o, 32); }
    const float e0 = p0 + b_cls[0], e1 = p1 + b_cls[1];

    const int lab = labels[b * TT + t];
    const int tag = (lab == -100) ? 0 : lab;
    const int mt  = (t == 0) ? 1 : mask[b * TT + t];

    if (t == 0) {
      alpha0 = crf_start[0] + e0;
      alpha1 = crf_start[1] + e1;
      num = crf_start[tag] + (tag ? e1 : e0);
    } else {
      const float n0 = lse2(alpha0 + t00, alpha1 + t10) + e0;
      const float n1 = lse2(alpha0 + t01, alpha1 + t11) + e1;
      if (mt) {
        alpha0 = n0; alpha1 = n1;
        num += crf_trans[prev_tag * 2 + tag] + (tag ? e1 : e0);
      }
    }
    prev_tag = tag;
  }

  if (lane == 0) {
    int cnt = 0;
    for (int t = 0; t < TT; ++t) cnt += (t == 0) ? 1 : (mask[b * TT + t] ? 1 : 0);
    const int lab = labels[b * TT + cnt - 1];
    const int ltag = (lab == -100) ? 0 : lab;
    num += crf_end[ltag];
    const float logZ = lse2(alpha0 + crf_end[0], alpha1 + crf_end[1]);
    res[b] = num - logZ;
  }
}

// ---------------- final combine ----------------

__global__ void final_kernel(const float* __restrict__ kd_part,
                             const float* __restrict__ res, float* __restrict__ out) {
  const int lane = threadIdx.x;
  float kd = 0.f;
  for (int i = lane; i < KD_BLOCKS; i += 32) kd += kd_part[i];
  float s = res[lane];
  for (int o = 16; o >= 1; o >>= 1) { kd += __shfl_xor(kd, o, 32); s += __shfl_xor(s, o, 32); }
  if (lane == 0) {
    const float span = -(s / (float)BB);
    const float kdm  = kd / (3.0f * BB * TT * DD);
    out[0] = 0.5f * span + 0.5f * kdm;
  }
}

// ---------------- host ----------------

extern "C" void kernel_launch(void* const* d_in, const int* in_sizes, int n_in,
                              void* d_out, int out_size, void* d_ws, size_t ws_size,
                              hipStream_t stream) {
  const float* top_hidden = (const float*)d_in[0];
  const float* s_feats    = (const float*)d_in[1];
  const float* t_feats    = (const float*)d_in[2];
  const float* W_ih_f = (const float*)d_in[3];
  const float* W_hh_f = (const float*)d_in[4];
  const float* b_ih_f = (const float*)d_in[5];
  const float* b_hh_f = (const float*)d_in[6];
  const float* W_ih_b = (const float*)d_in[7];
  const float* W_hh_b = (const float*)d_in[8];
  const float* b_ih_b = (const float*)d_in[9];
  const float* b_hh_b = (const float*)d_in[10];
  const float* W_cls  = (const float*)d_in[11];
  const float* b_cls  = (const float*)d_in[12];
  const float* crf_start = (const float*)d_in[13];
  const float* crf_end   = (const float*)d_in[14];
  const float* crf_trans = (const float*)d_in[15];
  const int*   labels = (const int*)d_in[16];
  const int*   mask   = (const int*)d_in[17];

  char* ws = (char*)d_ws;
  // workspace layout (bytes)
  const size_t OFF_KD   = 0;                                   // 1024 f32
  const size_t OFF_RES  = OFF_KD  + (size_t)KD_BLOCKS * 4;     // 32 f32
  const size_t OFF_XBF  = 8192;                                // bf16 x: 14336*768
  const size_t OFF_WIHF = OFF_XBF  + (size_t)MM * DD * 2;
  const size_t OFF_WIHB = OFF_WIHF + (size_t)G4 * DD * 2;
  const size_t OFF_WHHF = OFF_WIHB + (size_t)G4 * DD * 2;
  const size_t OFF_WHHB = OFF_WHHF + (size_t)G4 * HH * 2;
  const size_t OFF_GXF  = OFF_WHHB + (size_t)G4 * HH * 2;
  const size_t OFF_GXB  = OFF_GXF  + (size_t)MM * G4 * 4;
  const size_t OFF_HF   = OFF_GXB  + (size_t)MM * G4 * 4;
  const size_t OFF_HB   = OFF_HF   + (size_t)MM * HH * 4;

  float* kd_part = (float*)(ws + OFF_KD);
  float* res     = (float*)(ws + OFF_RES);
  u16* xbf  = (u16*)(ws + OFF_XBF);
  u16* wihf = (u16*)(ws + OFF_WIHF);
  u16* wihb = (u16*)(ws + OFF_WIHB);
  u16* whhf = (u16*)(ws + OFF_WHHF);
  u16* whhb = (u16*)(ws + OFF_WHHB);
  float* gxf = (float*)(ws + OFF_GXF);
  float* gxb = (float*)(ws + OFF_GXB);
  float* hf  = (float*)(ws + OFF_HF);
  float* hb  = (float*)(ws + OFF_HB);

  // 1) KD partial sums (streaming, memory bound)
  kd_kernel<<<KD_BLOCKS, 256, 0, stream>>>(s_feats, t_feats, kd_part,
                                           (size_t)3 * BB * TT * DD);

  // 2) f32 -> bf16 conversions
  cvt_bf16_kernel<<<2048, 256, 0, stream>>>(top_hidden, xbf, (size_t)MM * DD);
  cvt_bf16_kernel<<<512, 256, 0, stream>>>(W_ih_f, wihf, (size_t)G4 * DD);
  cvt_bf16_kernel<<<512, 256, 0, stream>>>(W_ih_b, wihb, (size_t)G4 * DD);
  cvt_bf16_kernel<<<256, 256, 0, stream>>>(W_hh_f, whhf, (size_t)G4 * HH);
  cvt_bf16_kernel<<<256, 256, 0, stream>>>(W_hh_b, whhb, (size_t)G4 * HH);

  // 3) gate pre-activations via WMMA: 14336 wave strips/dir, 8 waves/block
  gx_gemm_kernel<<<dim3(MM / 16 * 16 / 8, 1, 2), 256, 0, stream>>>(
      xbf, wihf, wihb, b_ih_f, b_hh_f, b_ih_b, b_hh_b, gxf, gxb);

  // 4) persistent LSTM scan, one WGP-resident workgroup per direction
  const size_t smem = (size_t)BB * G4 * 4 + (size_t)BB * HH * 4 + (size_t)BB * HH * 2;
  lstm_scan_kernel<<<2, 1024, smem, stream>>>(gxf, gxb, whhf, whhb, hf, hb);

  // 5) classifier + CRF per batch row
  crf_kernel<<<BB, 32, 0, stream>>>(hf, hb, W_cls, b_cls, crf_start, crf_end,
                                    crf_trans, labels, mask, res);

  // 6) final loss
  final_kernel<<<1, 32, 0, stream>>>(kd_part, res, (float*)d_out);
}
